// WindowAttention_63299228008579
// MI455X (gfx1250) — matrix operand order repaired and checked by reading, not compiled, not run on hardware
//
#include <hip/hip_runtime.h>
#include <hip/hip_bf16.h>

// Fused Swin window attention for gfx1250 (MI455X), wave32 + WMMA f16.
// One workgroup (128 threads = 4 waves) per window; one row-tile per wave in
// the GEMM phases, one head per wave in the attention phase.
// Weights are pre-converted to f16 once (tiny kernel) into d_ws.

typedef _Float16 half_t;
typedef __attribute__((ext_vector_type(16))) _Float16 v16h;
typedef __attribute__((ext_vector_type(8)))  _Float16 v8h;
typedef __attribute__((ext_vector_type(8)))  float    v8f;

namespace {
constexpr int EMBED = 128;
constexpr int NHEAD = 4;
constexpr int LTOK  = 49;   // tokens per window
constexpr int DK    = 32;   // head dim
constexpr int LPAD  = 64;   // padded token count (4 x 16 tiles)
constexpr int NWIN  = 1024; // windows per image
constexpr int WELEM = EMBED * EMBED;  // elements per weight matrix
}

__device__ __forceinline__ v8f wmma16(v16h a, v16h b, v8f c) {
  // v_wmma_f32_16x16x32_f16: D = A(16x32 f16) * B(32x16 f16) + C(16x16 f32)
  return __builtin_amdgcn_wmma_f32_16x16x32_f16(false, a, false, b,
                                                (short)0, c, false, false);
}

// ---- A-operand (16x32, row-major source, 16-bit):
// lane<16: row=lane,    elems = K{0..7, 16..23}
// lane>=16: row=lane-16, elems = K{8..15, 24..31}
__device__ __forceinline__ v16h lds_tileA(const half_t* base, int ld, int lane) {
  const int r  = lane & 15;
  const int kb = (lane & 16) ? 8 : 0;
  const half_t* p = base + r * ld + kb;
  v8h lo = *(const v8h*)(p);
  v8h hi = *(const v8h*)(p + 16);
  v16h t;
#pragma unroll
  for (int i = 0; i < 8; ++i) { t[i] = lo[i]; t[8 + i] = hi[i]; }
  return t;
}

// A-operand directly from global fp32 (rows = tokens; clamp padded rows)
__device__ __forceinline__ v16h g_tileA(const float* base, int ld, int lane, int rmax) {
  int r = lane & 15;
  if (r > rmax) r = rmax;                    // clamp OOB padded token rows
  const int kb = (lane & 16) ? 8 : 0;
  const float* p = base + r * ld + kb;
  v16h t;
#pragma unroll
  for (int i = 0; i < 8; ++i) {
    t[i]     = (half_t)p[i];
    t[8 + i] = (half_t)p[i + 16];
  }
  return t;
}

// ---- B-operand (32x16, column-contiguous source, 16-bit):
// lane<16: col=lane, elems = K{0..7, 16..23} ; lane>=16: col=lane-16, elems = K{8..15, 24..31}
__device__ __forceinline__ v16h lds_tileB(const half_t* base, int ld, int lane) {
  const int c  = lane & 15;
  const int kb = (lane & 16) ? 8 : 0;
  const half_t* p = base + c * ld + kb;
  v8h lo = *(const v8h*)(p);
  v8h hi = *(const v8h*)(p + 16);
  v16h t;
#pragma unroll
  for (int i = 0; i < 8; ++i) { t[i] = lo[i]; t[8 + i] = hi[i]; }
  return t;
}

// B-operand from pre-converted f16 weights in global: B = W^T, column n = row n of W
__device__ __forceinline__ v16h g_tileB_h(const half_t* base, int ld, int lane) {
  const int c  = lane & 15;
  const int kb = (lane & 16) ? 8 : 0;
  const half_t* p = base + c * ld + kb;
  v8h lo = *(const v8h*)(p);
  v8h hi = *(const v8h*)(p + 16);
  v16h t;
#pragma unroll
  for (int i = 0; i < 8; ++i) { t[i] = lo[i]; t[8 + i] = hi[i]; }
  return t;
}

union SmemPh23 {
  struct {
    half_t vt[NHEAD * DK * LPAD];   // V^T per head: [d][token]
    half_t p [NHEAD * 16 * LPAD];   // per-wave softmax row-block scratch
  } a;
  half_t o[LPAD * EMBED];           // concat-head attention output (phase 3)
};

// One-shot: convert the four 128x128 fp32 weight matrices to f16 in d_ws.
__global__ void __launch_bounds__(256)
convert_weights_kernel(const float* __restrict__ Wq, const float* __restrict__ Wk,
                       const float* __restrict__ Wv, const float* __restrict__ Wo,
                       half_t* __restrict__ out)
{
  const int i = blockIdx.x * blockDim.x + threadIdx.x;
  if (i < WELEM) {
    out[i]             = (half_t)Wq[i];
    out[WELEM + i]     = (half_t)Wk[i];
    out[2 * WELEM + i] = (half_t)Wv[i];
    out[3 * WELEM + i] = (half_t)Wo[i];
  }
}

__global__ void __launch_bounds__(128)
swin_window_attention_kernel(const float* __restrict__ q,
                             const float* __restrict__ k,
                             const float* __restrict__ v,
                             const float* __restrict__ mask,
                             const half_t* __restrict__ W16,   // [Wq|Wk|Wv|Wo] f16
                             const float* __restrict__ bo,
                             const float* __restrict__ rel_table,
                             float* __restrict__ out)
{
  __shared__ __align__(32) half_t  sQ[NHEAD * LPAD * DK];   // [h][token][d]
  __shared__ __align__(32) half_t  sK[NHEAD * LPAD * DK];   // [h][token][d]
  __shared__ __align__(32) SmemPh23 su;
  __shared__ float  sRel[169 * NHEAD];
  __shared__ half_t sMask[LTOK * LTOK];   // shift mask, exact in f16 (0 / -100)

  const int tid  = threadIdx.x;
  const int lane = tid & 31;
  const int wave = tid >> 5;
  const int b    = blockIdx.x;
  const int win  = b & (NWIN - 1);
  const float* maskw = mask + (size_t)win * LTOK * LTOK;
  const int cn = lane & 15;
  const int rl = (lane & 16) ? 8 : 0;

  // stage relative-position bias table + per-window shift mask (reused by all
  // 4 heads; mask values are exactly 0/-100 -> lossless in f16)
  for (int i = tid; i < 169 * NHEAD; i += blockDim.x) sRel[i] = rel_table[i];
  for (int i = tid; i < LTOK * LTOK; i += blockDim.x) sMask[i] = (half_t)maskw[i];

  // ---------------- Phase 1: projections Q,K,V  (Y = X @ W^T) ----------------
  // One token row-tile per wave; A-tiles loaded once, reused across 8 col-tiles.
  // B-tiles batched per col-tile so loads overlap the WMMA chain.
  const float* Xs[3] = {q, k, v};
  const int rt  = wave;                      // this wave's token row-tile
  const int rcl = (rt == 3) ? 0 : 15;        // clamp rows 49..63 -> row 48
#pragma unroll
  for (int pidx = 0; pidx < 3; ++pidx) {
    const float*  X = Xs[pidx] + (size_t)b * LTOK * EMBED;
    const half_t* W = W16 + pidx * WELEM;
    v16h a[4];
#pragma unroll
    for (int ks = 0; ks < 4; ++ks)
      a[ks] = g_tileA(X + rt * 16 * EMBED + ks * 32, EMBED, lane, rcl);
    for (int nt = 0; nt < 8; ++nt) {
      v16h bm[4];
#pragma unroll
      for (int ks = 0; ks < 4; ++ks)
        bm[ks] = g_tileB_h(W + nt * 16 * EMBED + ks * 32, EMBED, lane);
      v8f acc = {};
#pragma unroll
      for (int ks = 0; ks < 4; ++ks) acc = wmma16(a[ks], bm[ks], acc);
      const int h  = nt >> 1;
      const int d0 = (nt & 1) * 16;
#pragma unroll
      for (int i = 0; i < 8; ++i) {
        const int tok = rt * 16 + i + rl;
        const half_t val = (half_t)acc[i];
        if (pidx == 0)      sQ[(h * LPAD + tok) * DK + d0 + cn] = val;
        else if (pidx == 1) sK[(h * LPAD + tok) * DK + d0 + cn] = val;
        else                su.a.vt[(h * DK + d0 + cn) * LPAD + tok] = val;
      }
    }
  }
  __syncthreads();

  // ---------------- Phase 2: attention (one wave per head) ----------------
  const int h = wave;
  const half_t* sQh  = sQ + h * LPAD * DK;
  const half_t* sKh  = sK + h * LPAD * DK;
  const half_t* sVth = su.a.vt + h * DK * LPAD;
  half_t* sPw = su.a.p + wave * 16 * LPAD;
  const float scale = 0.17677669529663687f;    // 1/sqrt(32)

  // hoist row-tile-invariant B operands: K tiles (energy) and V^T tiles (AV)
  v16h bk[4], bv[2][2];
#pragma unroll
  for (int j = 0; j < 4; ++j) bk[j] = lds_tileB(sKh + j * 16 * DK, DK, lane);
#pragma unroll
  for (int t = 0; t < 2; ++t)
#pragma unroll
    for (int ks = 0; ks < 2; ++ks)
      bv[t][ks] = lds_tileB(sVth + t * 16 * LPAD + ks * 32, LPAD, lane);

  v8f oacc[4][2];
#pragma unroll 1
  for (int rtt = 0; rtt < 4; ++rtt) {
    // energy tiles: one WMMA each (K = d = 32)
    v16h aq = lds_tileA(sQh + rtt * 16 * DK, DK, lane);
    v8f e[4];
#pragma unroll
    for (int j = 0; j < 4; ++j) {
      v8f z = {};
      e[j] = wmma16(aq, bk[j], z);
    }
    // scale + relative bias + shift mask, then row softmax; P -> LDS f16
#pragma unroll
    for (int i = 0; i < 8; ++i) {
      const int row = rtt * 16 + i + rl;
      float ev[4];
#pragma unroll
      for (int j = 0; j < 4; ++j) {
        const int col = j * 16 + cn;
        float val = -1.0e30f;
        if (row < LTOK && col < LTOK) {
          const int dh = row / 7 - col / 7 + 6;
          const int dw = row % 7 - col % 7 + 6;
          val = e[j][i] * scale + sRel[(dh * 13 + dw) * NHEAD + h]
              + (float)sMask[row * LTOK + col];
        }
        ev[j] = val;
      }
      float mx = fmaxf(fmaxf(ev[0], ev[1]), fmaxf(ev[2], ev[3]));
#pragma unroll
      for (int off = 8; off >= 1; off >>= 1) mx = fmaxf(mx, __shfl_xor(mx, off));
      float s = 0.f;
#pragma unroll
      for (int j = 0; j < 4; ++j) { ev[j] = __expf(ev[j] - mx); s += ev[j]; }
#pragma unroll
      for (int off = 8; off >= 1; off >>= 1) s += __shfl_xor(s, off);
      const float inv = 1.0f / s;
#pragma unroll
      for (int j = 0; j < 4; ++j)
        sPw[(i + rl) * LPAD + j * 16 + cn] = (half_t)(ev[j] * inv);
    }
    // AV for this row tile: out[16,32] = P[16,64] @ V[64,32]  (2 k-steps)
#pragma unroll
    for (int t = 0; t < 2; ++t) {
      v8f acc = {};
#pragma unroll
      for (int ks = 0; ks < 2; ++ks)
        acc = wmma16(lds_tileA(sPw + ks * 32, LPAD, lane), bv[t][ks], acc);
      oacc[rtt][t] = acc;
    }
  }
  __syncthreads();   // everyone done with sVt/sP -> union now becomes sO

  // write concat-head attention output to LDS (f16)
#pragma unroll
  for (int rtt = 0; rtt < 4; ++rtt)
#pragma unroll
    for (int t = 0; t < 2; ++t)
#pragma unroll
      for (int i = 0; i < 8; ++i) {
        const int row = rtt * 16 + i + rl;
        const int col = h * DK + t * 16 + cn;
        su.o[row * EMBED + col] = (half_t)oacc[rtt][t][i];
      }
  __syncthreads();

  // ---------------- Phase 3: output projection + bias ----------------
  // One row-tile per wave again; A-tiles (attention output) hoisted,
  // weight B-tiles batched ahead of the WMMA chain.
  float* outb = out + (size_t)b * LTOK * EMBED;
  const half_t* Wo16 = W16 + 3 * WELEM;
  v16h ao[4];
#pragma unroll
  for (int ks = 0; ks < 4; ++ks)
    ao[ks] = lds_tileA(su.o + rt * 16 * EMBED + ks * 32, EMBED, lane);
  for (int nt = 0; nt < 8; ++nt) {
    v16h bm[4];
#pragma unroll
    for (int ks = 0; ks < 4; ++ks)
      bm[ks] = g_tileB_h(Wo16 + nt * 16 * EMBED + ks * 32, EMBED, lane);
    v8f acc = {};
#pragma unroll
    for (int ks = 0; ks < 4; ++ks) acc = wmma16(ao[ks], bm[ks], acc);
    const int col = nt * 16 + cn;
    const float bvv = bo[col];
#pragma unroll
    for (int i = 0; i < 8; ++i) {
      const int row = rt * 16 + i + rl;
      if (row < LTOK) outb[(size_t)row * EMBED + col] = acc[i] + bvv;
    }
  }
}

extern "C" void kernel_launch(void* const* d_in, const int* in_sizes, int n_in,
                              void* d_out, int out_size, void* d_ws, size_t ws_size,
                              hipStream_t stream) {
  (void)n_in; (void)out_size; (void)ws_size;
  const float* q         = (const float*)d_in[0];
  const float* k         = (const float*)d_in[1];
  const float* v         = (const float*)d_in[2];
  const float* mask      = (const float*)d_in[3];
  const float* Wq        = (const float*)d_in[4];
  const float* Wk        = (const float*)d_in[5];
  const float* Wv        = (const float*)d_in[6];
  const float* Wo        = (const float*)d_in[7];
  const float* bo        = (const float*)d_in[8];
  const float* rel_table = (const float*)d_in[9];
  float* out   = (float*)d_out;
  half_t* W16  = (half_t*)d_ws;   // 4 * 128*128 f16 = 128 KB

  // 1) pre-convert weights to f16 (block-invariant, tiny)
  convert_weights_kernel<<<(WELEM + 255) / 256, 256, 0, stream>>>(Wq, Wk, Wv, Wo, W16);

  // 2) fused attention, one workgroup per window
  const int B = in_sizes[0] / (LTOK * EMBED);   // 8192 windows
  swin_window_attention_kernel<<<B, 128, 0, stream>>>(
      q, k, v, mask, W16, bo, rel_table, out);
}
